// WindowAttention3D_22110491640506
// MI455X (gfx1250) — matrix operand order repaired and checked by reading, not compiled, not run on hardware
//
#include <hip/hip_runtime.h>
#include <hip/hip_bf16.h>

typedef _Float16 f16;
typedef __attribute__((ext_vector_type(16))) _Float16 v16h;
typedef __attribute__((ext_vector_type(8)))  float    v8f;

#define DIM    192
#define HEADS  6
#define NTOK   98
#define NPAD   112
#define HD     32
#define BATCH  1536
#define ROWS   (BATCH * NTOK)      // 150528
#define ROWT   (ROWS / 16)         // 9408 row tiles
#define QKVC   (3 * DIM)           // 576

// ---- WMMA operand builders (16x16x32 f16) -------------------------------
// A (16x32): lane L -> row M = L%16, khalf = L/16.
//   elements 0..7  = K[khalf*8 + 0..7], elements 8..15 = K[16 + khalf*8 + 0..7]
static __device__ __forceinline__ v16h load_A(const f16* p /* row base + khalf*8 */) {
  v16h a;
#pragma unroll
  for (int e = 0; e < 8; ++e) a[e] = p[e];
#pragma unroll
  for (int e = 0; e < 8; ++e) a[8 + e] = p[16 + e];
  return a;
}
// B (32x16): lane L -> col N = L%16, khalf = L/16.
//   elements 0..7  = K[khalf*8 + 0..7], elements 8..15 = K[16 + khalf*8 + 0..7]
static __device__ __forceinline__ v16h load_B(const f16* p /* col base + khalf*8 */) {
  v16h b;
#pragma unroll
  for (int e = 0; e < 8; ++e) b[e] = p[e];
#pragma unroll
  for (int e = 0; e < 8; ++e) b[8 + e] = p[16 + e];
  return b;
}

// ---- prep: f32->f16 weights; fuse rel_bias gather into (H,98,98) table ---
__global__ void __launch_bounds__(256) prep_kernel(
    const float* __restrict__ qkv_w, const float* __restrict__ proj_w,
    const float* __restrict__ rel_bias, const int* __restrict__ rel_idx,
    f16* __restrict__ whq, f16* __restrict__ whp, float* __restrict__ bias2) {
  int tid = blockIdx.x * 256 + threadIdx.x;
  if (tid < QKVC * DIM) { whq[tid] = (f16)qkv_w[tid]; return; }
  tid -= QKVC * DIM;
  if (tid < DIM * DIM) { whp[tid] = (f16)proj_w[tid]; return; }
  tid -= DIM * DIM;
  if (tid < HEADS * NTOK * NTOK) {
    int h  = tid / (NTOK * NTOK);
    int ij = tid % (NTOK * NTOK);
    bias2[tid] = rel_bias[rel_idx[ij] * HEADS + h];
  }
}

// ---- QKV GEMM: (150528,192) x (192,576)^T + bias -> Q,K (f16, [b,h,t,d]) and V^T (f16, [b,h,d,t])
__global__ void __launch_bounds__(256) qkv_kernel(
    const float* __restrict__ x, const float* __restrict__ qkv_b,
    const f16* __restrict__ whq,
    f16* __restrict__ Qb, f16* __restrict__ Kb, f16* __restrict__ Vt) {
  __shared__ __align__(16) f16 xs[16 * DIM];
  const int  tid     = threadIdx.x;
  const long rowbase = (long)blockIdx.x * 16;
  const float* xg = x + rowbase * DIM;
#pragma unroll
  for (int f = tid; f < 16 * DIM; f += 256) xs[f] = (f16)xg[f];
  __syncthreads();

  const int wave = tid >> 5, lane = tid & 31;
  const int kh = lane >> 4, ln = lane & 15;
  const float scale = 0.17677669529663687f;  // 32^-0.5

  for (int ct = wave; ct < QKVC / 16; ct += 8) {   // 36 col tiles over 8 waves
    const int col = ct * 16 + ln;
    const float cb = qkv_b[col];
    v8f c;
#pragma unroll
    for (int i = 0; i < 8; ++i) c[i] = cb;
#pragma unroll
    for (int kc = 0; kc < 6; ++kc) {               // K = 192 = 6 x 32
      v16h a  = load_A(&xs[ln * DIM + kc * 32 + kh * 8]);
      v16h bv = load_B(&whq[(size_t)col * DIM + kc * 32 + kh * 8]);
      c = __builtin_amdgcn_wmma_f32_16x16x32_f16(false, a, false, bv, (short)0, c, false, false);
    }
    // `which` is tile-uniform (16-col tiles never straddle a 192 boundary):
    // readfirstlane makes the 3-way split a scalar branch, not an EXEC dance.
    const int which = __builtin_amdgcn_readfirstlane(ct * 16) / DIM;
    const int cc = col % DIM;
    const int hh = cc >> 5, dd = cc & 31;
    // 8 consecutive rows: one div-by-98, then increment with wrap
    const long r0 = rowbase + 8 * kh;
    int bb = (int)(r0 / NTOK);
    int t  = (int)(r0 - (long)bb * NTOK);
    if (which == 0) {
#pragma unroll
      for (int i = 0; i < 8; ++i) {
        Qb[(((size_t)bb * HEADS + hh) * NTOK + t) * HD + dd] = (f16)(c[i] * scale);
        if (++t == NTOK) { t = 0; ++bb; }
      }
    } else if (which == 1) {
#pragma unroll
      for (int i = 0; i < 8; ++i) {
        Kb[(((size_t)bb * HEADS + hh) * NTOK + t) * HD + dd] = (f16)c[i];
        if (++t == NTOK) { t = 0; ++bb; }
      }
    } else {
#pragma unroll
      for (int i = 0; i < 8; ++i) {
        Vt[(((size_t)bb * HEADS + hh) * HD + dd) * NTOK + t] = (f16)c[i];  // transposed for PV B-operand
        if (++t == NTOK) { t = 0; ++bb; }
      }
    }
  }
}

// ---- fused attention: S = QK^T + bias + mask; softmax; O = P V --------------
__global__ void __launch_bounds__(256) attn_kernel(
    const float* __restrict__ mask, const float* __restrict__ bias2,
    const f16* __restrict__ Qb, const f16* __restrict__ Kb, const f16* __restrict__ Vt,
    f16* __restrict__ aout) {
  __shared__ __align__(16) f16 Qs[NPAD * HD];       // 7 KB
  __shared__ __align__(16) f16 Ks[NPAD * HD];       // 7 KB
  __shared__ __align__(16) f16 Vs[HD * NPAD];       // 7 KB  (V^T: [d][t])
  __shared__ __align__(16) f16 Ps[7 * 16 * 128];    // 28 KB (wave-private P tiles, K padded to 128)

  const int b    = blockIdx.x;
  const int tid  = threadIdx.x;
  const int wave = tid >> 5, lane = tid & 31;
  const int kh   = lane >> 4, ln = lane & 15;
  const float* maskg = mask + (size_t)b * NTOK * NTOK;

  for (int h = 0; h < HEADS; ++h) {
    const size_t bh = (size_t)b * HEADS + h;
    const f16* Qg = Qb + bh * NTOK * HD;
    const f16* Kg = Kb + bh * NTOK * HD;
    const f16* Vg = Vt + bh * HD * NTOK;
    // stage Q,K,V^T padded to 112 rows/cols (zeros)
    for (int f = tid; f < NPAD * HD; f += 256) {
      int t = f >> 5;
      f16 q = (t < NTOK) ? Qg[f] : (f16)0;
      f16 k = (t < NTOK) ? Kg[f] : (f16)0;
      Qs[f] = q; Ks[f] = k;
    }
    for (int f = tid; f < HD * NPAD; f += 256) {
      int d = f / NPAD, t = f - d * NPAD;
      Vs[f] = (t < NTOK) ? Vg[d * NTOK + t] : (f16)0;
    }
    // prefetch next head's tiles (global_prefetch_b8) to overlap with compute
    if (h + 1 < HEADS) {
      const f16* Qn = Qg + NTOK * HD;   // (bh+1) regions are contiguous
      const f16* Kn = Kg + NTOK * HD;
      const f16* Vn = Vg + HD * NTOK;
      for (int f = tid * 32; f < NTOK * HD; f += 256 * 32) {   // one 64B line / lane
        __builtin_prefetch(Qn + f, 0, 3);
        __builtin_prefetch(Kn + f, 0, 3);
        __builtin_prefetch(Vn + f, 0, 3);
      }
    }
    __syncthreads();

    if (wave < 7) {                                  // 7 row-tiles of S
      const int w = wave;
      const float* b2g = bias2 + (size_t)h * NTOK * NTOK;
      v8f sc[7];
      // accumulator starts at bias + mask (padded cols forced to -inf)
#pragma unroll
      for (int ct = 0; ct < 7; ++ct) {
        const int j = ct * 16 + ln;
#pragma unroll
        for (int i = 0; i < 8; ++i) {
          const int ti = w * 16 + i + 8 * kh;
          float cv;
          if (j >= NTOK)      cv = -1e30f;
          else if (ti < NTOK) cv = maskg[ti * NTOK + j] + b2g[ti * NTOK + j];
          else                cv = 0.0f;
          sc[ct][i] = cv;
        }
      }
      // S += Q K^T : one 16x16x32 WMMA per tile (hd == 32)
      v16h a = load_A(&Qs[(w * 16 + ln) * HD + kh * 8]);
#pragma unroll
      for (int ct = 0; ct < 7; ++ct) {
        v16h bv = load_B(&Ks[(ct * 16 + ln) * HD + kh * 8]);
        sc[ct] = __builtin_amdgcn_wmma_f32_16x16x32_f16(false, a, false, bv, (short)0, sc[ct], false, false);
      }
      // register-resident softmax: reduce over 7 tiles then across 16-lane column group
      float inv[8];
#pragma unroll
      for (int i = 0; i < 8; ++i) {
        float m = sc[0][i];
#pragma unroll
        for (int ct = 1; ct < 7; ++ct) m = fmaxf(m, sc[ct][i]);
#pragma unroll
        for (int xm = 1; xm < 16; xm <<= 1) m = fmaxf(m, __shfl_xor(m, xm, 32));
        float s = 0.0f;
#pragma unroll
        for (int ct = 0; ct < 7; ++ct) { float e = __expf(sc[ct][i] - m); sc[ct][i] = e; s += e; }
#pragma unroll
        for (int xm = 1; xm < 16; xm <<= 1) s += __shfl_xor(s, xm, 32);
        inv[i] = 1.0f / s;
      }
      // C/D layout -> A layout via wave-private LDS (f16), zero K-pad 112..127
      f16* Pw = &Ps[w * 16 * 128];
#pragma unroll
      for (int ct = 0; ct < 7; ++ct)
#pragma unroll
        for (int i = 0; i < 8; ++i)
          Pw[(i + 8 * kh) * 128 + ct * 16 + ln] = (f16)(sc[ct][i] * inv[i]);
#pragma unroll
      for (int i = 0; i < 8; ++i) Pw[(i + 8 * kh) * 128 + 112 + ln] = (f16)0;

      // O = P V : 2 d-tiles x 4 K-chunks of 32
#pragma unroll
      for (int dT = 0; dT < 2; ++dT) {
        v8f acc = {};
#pragma unroll
        for (int kc = 0; kc < 4; ++kc) {
          v16h pa = load_A(&Pw[ln * 128 + kc * 32 + kh * 8]);
          v16h vb = load_B(&Vs[(dT * 16 + ln) * NPAD + kc * 32 + kh * 8]);
          acc = __builtin_amdgcn_wmma_f32_16x16x32_f16(false, pa, false, vb, (short)0, acc, false, false);
        }
#pragma unroll
        for (int i = 0; i < 8; ++i) {
          const int t = w * 16 + i + 8 * kh;
          if (t < NTOK)
            aout[((size_t)b * NTOK + t) * DIM + h * HD + dT * 16 + ln] = (f16)acc[i];
        }
      }
    }
    __syncthreads();
  }
}

// ---- proj GEMM: (150528,192) x (192,192)^T + bias -> f32 out -------------
__global__ void __launch_bounds__(256) proj_kernel(
    const f16* __restrict__ aout, const float* __restrict__ proj_b,
    const f16* __restrict__ whp, float* __restrict__ out) {
  __shared__ __align__(16) f16 as[16 * DIM];
  const int  tid     = threadIdx.x;
  const long rowbase = (long)blockIdx.x * 16;
  const f16* ag = aout + rowbase * DIM;
  for (int f = tid; f < 16 * DIM; f += 256) as[f] = ag[f];
  __syncthreads();

  const int wave = tid >> 5, lane = tid & 31;
  const int kh = lane >> 4, ln = lane & 15;
  for (int ct = wave; ct < DIM / 16; ct += 8) {     // 12 col tiles
    const int col = ct * 16 + ln;
    const float cb = proj_b[col];
    v8f c;
#pragma unroll
    for (int i = 0; i < 8; ++i) c[i] = cb;
#pragma unroll
    for (int kc = 0; kc < 6; ++kc) {
      v16h a  = load_A(&as[ln * DIM + kc * 32 + kh * 8]);
      v16h bv = load_B(&whp[(size_t)col * DIM + kc * 32 + kh * 8]);
      c = __builtin_amdgcn_wmma_f32_16x16x32_f16(false, a, false, bv, (short)0, c, false, false);
    }
#pragma unroll
    for (int i = 0; i < 8; ++i)
      out[(rowbase + i + 8 * kh) * DIM + col] = c[i];
  }
}

// ---- host glue -----------------------------------------------------------
static inline size_t align_up(size_t v, size_t a) { return (v + a - 1) & ~(a - 1); }

extern "C" void kernel_launch(void* const* d_in, const int* in_sizes, int n_in,
                              void* d_out, int out_size, void* d_ws, size_t ws_size,
                              hipStream_t stream) {
  (void)in_sizes; (void)n_in; (void)out_size; (void)ws_size;
  const float* x        = (const float*)d_in[0];
  const float* mask     = (const float*)d_in[1];
  const float* qkv_w    = (const float*)d_in[2];
  const float* qkv_b    = (const float*)d_in[3];
  const float* proj_w   = (const float*)d_in[4];
  const float* proj_b   = (const float*)d_in[5];
  const float* rel_bias = (const float*)d_in[6];
  const int*   rel_idx  = (const int*)d_in[7];
  float* out = (float*)d_out;

  char* ws = (char*)d_ws;
  size_t off = 0;
  f16*   whq   = (f16*)(ws + off); off = align_up(off + (size_t)QKVC * DIM * 2, 512);
  f16*   whp   = (f16*)(ws + off); off = align_up(off + (size_t)DIM * DIM * 2, 512);
  float* bias2 = (float*)(ws + off); off = align_up(off + (size_t)HEADS * NTOK * NTOK * 4, 512);
  const size_t qk_elems = (size_t)BATCH * HEADS * NTOK * HD;   // 28,901,376
  f16* Qb   = (f16*)(ws + off); off = align_up(off + qk_elems * 2, 512);
  f16* Kb   = (f16*)(ws + off); off = align_up(off + qk_elems * 2, 512);
  f16* Vt   = (f16*)(ws + off); off = align_up(off + qk_elems * 2, 512);
  f16* aout = (f16*)(ws + off); off = align_up(off + (size_t)ROWS * DIM * 2, 512);

  const int prep_total = QKVC * DIM + DIM * DIM + HEADS * NTOK * NTOK;
  prep_kernel<<<(prep_total + 255) / 256, 256, 0, stream>>>(qkv_w, proj_w, rel_bias, rel_idx,
                                                            whq, whp, bias2);
  qkv_kernel<<<ROWT, 256, 0, stream>>>(x, qkv_b, whq, Qb, Kb, Vt);
  attn_kernel<<<BATCH, 256, 0, stream>>>(mask, bias2, Qb, Kb, Vt, aout);
  proj_kernel<<<ROWT, 256, 0, stream>>>(aout, proj_b, whp, out);
}
